// PPN_74972949119321
// MI455X (gfx1250) — matrix operand order repaired and checked by reading, not compiled, not run on hardware
//
#include <hip/hip_runtime.h>
#include <hip/hip_bf16.h>

typedef __attribute__((ext_vector_type(16))) _Float16 v16h;
typedef __attribute__((ext_vector_type(8)))  _Float16 v8h;
typedef __attribute__((ext_vector_type(8)))  float    v8f;
typedef __attribute__((ext_vector_type(4)))  unsigned u32x4;
typedef __attribute__((ext_vector_type(8)))  int      i32x8;
typedef __attribute__((ext_vector_type(4)))  int      i32x4;

// ---------------------------------------------------------------------------
// Tensor Data Mover: 1-D async copy global -> LDS (nelem8 x 8 bytes).
// D# built per CDNA5 ISA (group0/group1 bitfields, type=2).
// Wave-level op (EXEC ignored); caller must be wave-uniform and pair with
// s_wait_tensorcnt before publishing via barrier.
// ---------------------------------------------------------------------------
__device__ __forceinline__ void tdm_load_1d(unsigned lds_off, const void* gptr,
                                            unsigned nelem8) {
    unsigned long long va = (unsigned long long)gptr;
    u32x4 g0;
    g0[0] = 1u;                                            // count=1, user mode
    g0[1] = lds_off;                                       // lds_addr (bytes)
    g0[2] = (unsigned)va;                                  // global_addr[31:0]
    g0[3] = (unsigned)((va >> 32) & 0x01FFFFFFu)           // global_addr[56:32]
            | 0x80000000u;                                 // type=2 (bits 127:126)
    i32x8 g1;
    g1[0] = 0x00030000;                                    // data_size=3 (8B), no mask
    g1[1] = (int)((nelem8 & 0xFFFFu) << 16);               // tensor_dim0[15:0]
    g1[2] = (int)((nelem8 >> 16) | (1u << 16));            // tensor_dim0[31:16] | tensor_dim1=1
    g1[3] = (int)(nelem8 << 16);                           // tile_dim0
    g1[4] = 0;                                             // tile_dim1/2 unused
    g1[5] = (int)nelem8;                                   // tensor_dim0_stride
    g1[6] = 0;
    g1[7] = 0;
    i32x4 g2 = {0, 0, 0, 0};
    i32x4 g3 = {0, 0, 0, 0};
#if __clang_major__ >= 23
    i32x8 g4 = {0, 0, 0, 0, 0, 0, 0, 0};
    __builtin_amdgcn_tensor_load_to_lds(g0, g1, g2, g3, g4, 0);
#else
    __builtin_amdgcn_tensor_load_to_lds(g0, g1, g2, g3, 0);
#endif
}

// ---------------------------------------------------------------------------
// Repack W[27][Cin][Cout] (f32) into WMMA B-fragment layout (f16), zero-padded.
//   elem j of lane L of frag (tap,kb,nb):  K = kb*32+(L>>4)*16+j, n = nb*16+(L&15)
// ---------------------------------------------------------------------------
__global__ void repack_weights(const float* __restrict__ W, _Float16* __restrict__ Bfrag,
                               int Cin, int Cout, int KB, int NB, long total) {
    long t = (long)blockIdx.x * blockDim.x + threadIdx.x;
    if (t >= total) return;
    int j    = (int)(t & 15);
    int lane = (int)((t >> 4) & 31);
    long r   = t >> 9;
    int nb   = (int)(r % NB); r /= NB;
    int kb   = (int)(r % KB); r /= KB;
    int tap  = (int)r;
    int K = kb * 32 + (lane >> 4) * 16 + j;
    int n = nb * 16 + (lane & 15);
    float v = (K < Cin && n < Cout) ? W[((long)tap * Cin + K) * Cout + n] : 0.0f;
    Bfrag[t] = (_Float16)v;
}

// Concatenate W3p|W3s|W3t -> Wcat[27][16][10] (f32)
__global__ void concat_heads(const float* __restrict__ Wp, const float* __restrict__ Ws,
                             const float* __restrict__ Wt, float* __restrict__ Wcat) {
    int t = blockIdx.x * blockDim.x + threadIdx.x;
    if (t >= 27 * 16 * 10) return;
    int d = t % 10;
    int c = (t / 10) % 16;
    int tap = t / 160;
    float v;
    if (d < 3)      v = Wp[((long)tap * 16 + c) * 3 + d];
    else if (d < 5) v = Ws[((long)tap * 16 + c) * 2 + (d - 3)];
    else            v = Wt[((long)tap * 16 + c) * 5 + (d - 5)];
    Wcat[t] = v;
}

// feat[N][Cin] (f32) -> out[(N+1)][Kpad] (f16), channel pad + zero sentinel row.
__global__ void pad_features(const float* __restrict__ feat, _Float16* __restrict__ out,
                             int N, int Cin, int Kpad) {
    long t = (long)blockIdx.x * blockDim.x + threadIdx.x;
    long total = (long)(N + 1) * Kpad;
    if (t >= total) return;
    int  c   = (int)(t % Kpad);
    long row = t / Kpad;
    out[t] = (_Float16)((row < N && c < Cin) ? feat[row * Cin + c] : 0.0f);
}

// out[(N+1)][Kpad] (f16) = feat[N][Cin] * mask[parent[row]], zero-padded.
__global__ void gate_pad(const float* __restrict__ feat, const float* __restrict__ mask,
                         const int* __restrict__ parent, _Float16* __restrict__ out,
                         int N, int Cin, int Kpad) {
    long t = (long)blockIdx.x * blockDim.x + threadIdx.x;
    long total = (long)(N + 1) * Kpad;
    if (t >= total) return;
    int  c   = (int)(t % Kpad);
    long row = t / Kpad;
    float v = 0.0f;
    if (row < N && c < Cin) v = feat[row * Cin + c] * mask[parent[row]];
    out[t] = (_Float16)v;
}

// mask = (softmax(s)[:,1] > 0.8); also ppn = [coords(f32 x4), s(x2)] and mask out.
__global__ void mask_ppn(const float* __restrict__ s, const int* __restrict__ coords,
                         float* __restrict__ mask_ws, float* __restrict__ ppn_out,
                         float* __restrict__ mask_out, int N) {
    int i = blockIdx.x * blockDim.x + threadIdx.x;
    if (i >= N) return;
    float a = s[i * 2 + 0], b = s[i * 2 + 1];
    float p = 1.0f / (1.0f + __expf(a - b));
    float m = (p > 0.8f) ? 1.0f : 0.0f;
    mask_ws[i]  = m;
    mask_out[i] = m;
    ppn_out[i * 6 + 0] = (float)coords[i * 4 + 0];
    ppn_out[i * 6 + 1] = (float)coords[i * 4 + 1];
    ppn_out[i * 6 + 2] = (float)coords[i * 4 + 2];
    ppn_out[i * 6 + 3] = (float)coords[i * 4 + 3];
    ppn_out[i * 6 + 4] = a;
    ppn_out[i * 6 + 5] = b;
}

// ---------------------------------------------------------------------------
// Submanifold conv as gathered GEMM.
//  - One workgroup = 8 waves; each wave owns MT row-tiles (16 rows each).
//  - Per tap k, the per-tap B-fragment block (KB*NB KiB) is DMA'd into LDS by
//    the TDM (wave 0), double-buffered across taps; all waves consume it via
//    ds_load. A rows are gathered (f16, sentinel zero row N), so rows >= N
//    compute exact zeros.
//  - __launch_bounds__(256, 2): an 8-wave WG occupies 2 waves/SIMD anyway, so
//    budgeting registers for 2 waves/EU costs no occupancy and removes the
//    accumulator spills seen at the default budget.
// ---------------------------------------------------------------------------
template<int KB, int NB, int MT, typename OutT>
__global__ __launch_bounds__(256, 2)
void subconv_gemm(const _Float16* __restrict__ Apad,     // (N+1) x (KB*32) f16
                  const int*      __restrict__ nbr,      // N x 27
                  const _Float16* __restrict__ Bfrag,    // 27 frag blocks
                  OutT* __restrict__ out, int ldout, int coutStore,
                  int N, int Nstore) {
    __shared__ __align__(16) _Float16 smem[2][KB * NB * 512];

    const int lane  = threadIdx.x & 31;
    const int wave  = threadIdx.x >> 5;
    const int nw    = blockDim.x >> 5;
    const int tile0 = (blockIdx.x * nw + wave) * MT;
    const int m     = lane & 15;
    const int half  = lane >> 4;
    const int Kpad  = KB * 32;
    const unsigned tapBytes = (unsigned)(KB * NB * 1024);   // per-tap fragment bytes

    v8f acc[MT][NB];
    const v8f vzero = {0.f, 0.f, 0.f, 0.f, 0.f, 0.f, 0.f, 0.f};
#pragma unroll
    for (int t = 0; t < MT; ++t)
#pragma unroll
        for (int i = 0; i < NB; ++i) acc[t][i] = vzero;

    // Prologue: DMA tap 0 fragments into buffer 0.
    if (wave == 0) {
        tdm_load_1d((unsigned)(unsigned long long)&smem[0][0], Bfrag, tapBytes >> 3);
        __builtin_amdgcn_s_wait_tensorcnt(0);
    }
    __syncthreads();

    for (int k = 0; k < 27; ++k) {
        const int cur = k & 1;
        // Kick DMA for next tap into the other buffer (overlaps compute).
        if (wave == 0 && k + 1 < 27) {
            tdm_load_1d((unsigned)(unsigned long long)&smem[cur ^ 1][0],
                        Bfrag + (long)(k + 1) * (tapBytes >> 1), tapBytes >> 3);
        }

        // Gather A rows for this tap (sentinel row N when OOB/missing).
        const _Float16* arow[MT];
#pragma unroll
        for (int t = 0; t < MT; ++t) {
            int myrow = (tile0 + t) * 16 + m;
            int gidx = (myrow < N) ? nbr[(long)myrow * 27 + k] : N;
            arow[t] = Apad + (long)gidx * Kpad;
        }

#pragma unroll
        for (int kb = 0; kb < KB; ++kb) {
            // A fragment (contiguous k order, matches B repack):
            //   elems 0..15 -> K = kb*32 + half*16 + j
            v16h afrag[MT];
#pragma unroll
            for (int t = 0; t < MT; ++t) {
                afrag[t] = *(const v16h*)(arow[t] + kb * 32 + half * 16);
            }
#pragma unroll
            for (int nb = 0; nb < NB; ++nb) {
                const _Float16* bp = &smem[cur][((kb * NB + nb) * 32 + lane) * 16];
                v16h bfrag = *(const v16h*)bp;
#pragma unroll
                for (int t = 0; t < MT; ++t) {
                    acc[t][nb] = __builtin_amdgcn_wmma_f32_16x16x32_f16(
                        false, afrag[t], false, bfrag, (short)0, acc[t][nb], false, false);
                }
            }
        }

        if (wave == 0 && k + 1 < 27) __builtin_amdgcn_s_wait_tensorcnt(0);
        __syncthreads();    // publish next buffer / retire reads of current one
    }

    // Store per f32 C/D layout: col = lane&15, row(reg r) = r + half*8.
#pragma unroll
    for (int t = 0; t < MT; ++t) {
        const int rowBase = (tile0 + t) * 16 + half * 8;
#pragma unroll
        for (int nb = 0; nb < NB; ++nb) {
            const int col = nb * 16 + m;
#pragma unroll
            for (int r = 0; r < 8; ++r) {
                int row = rowBase + r;
                if (row < Nstore && col < coutStore)
                    out[(long)row * ldout + col] = (OutT)acc[t][nb][r];
            }
        }
    }
}

// ---------------------------------------------------------------------------

static inline int ceil_div(long a, long b) { return (int)((a + b - 1) / b); }

extern "C" void kernel_launch(void* const* d_in, const int* in_sizes, int n_in,
                              void* d_out, int out_size, void* d_ws, size_t ws_size,
                              hipStream_t stream) {
    const float* feat1 = (const float*)d_in[0];
    const float* feat2 = (const float*)d_in[1];
    const float* feat3 = (const float*)d_in[2];
    const float* W1    = (const float*)d_in[3];
    const float* W1s   = (const float*)d_in[4];
    const float* W2    = (const float*)d_in[5];
    const float* W2s   = (const float*)d_in[6];
    const float* W3    = (const float*)d_in[7];
    const float* W3p   = (const float*)d_in[8];
    const float* W3s   = (const float*)d_in[9];
    const float* W3t   = (const float*)d_in[10];
    const int* nbr1    = (const int*)d_in[11];
    const int* nbr2    = (const int*)d_in[12];
    const int* nbr3    = (const int*)d_in[13];
    const int* parent2 = (const int*)d_in[14];
    const int* parent3 = (const int*)d_in[15];
    const int* coords1 = (const int*)d_in[16];
    const int* coords2 = (const int*)d_in[17];

    const int N1 = in_sizes[0] / 80;
    const int N2 = in_sizes[1] / 48;
    const int N3 = in_sizes[2] / 16;

    // Output regions (reference return order)
    float* out_points = (float*)d_out;                    // [N3,10]
    float* out_ppn1   = out_points + (long)N3 * 10;       // [N1,6]
    float* out_ppn2   = out_ppn1 + (long)N1 * 6;          // [N2,6]
    float* out_mask1  = out_ppn2 + (long)N2 * 6;          // [N1]
    float* out_mask2  = out_mask1 + N1;                   // [N2]

    // Workspace carve-up (256B aligned)
    char* ws = (char*)d_ws;
    size_t off = 0;
    auto alloc = [&](size_t bytes) -> char* {
        char* p = ws + off;
        off = (off + bytes + 255) & ~(size_t)255;
        return p;
    };
    _Float16* A1pad = (_Float16*)alloc((size_t)(N1 + 1) * 96 * 2);   // feat1 pad (Kpad=96)
    _Float16* Ypad  = (_Float16*)alloc((size_t)(N1 + 1) * 96 * 2);
    float*    s1    = (float*)alloc((size_t)N1 * 2 * 4);
    float*    mask1 = (float*)alloc((size_t)N1 * 4);
    _Float16* X2pad = (_Float16*)alloc((size_t)(N2 + 1) * 64 * 2);   // Kpad=64
    _Float16* Y2pad = (_Float16*)alloc((size_t)(N2 + 1) * 64 * 2);
    float*    s2    = (float*)alloc((size_t)N2 * 2 * 4);
    float*    mask2 = (float*)alloc((size_t)N2 * 4);
    _Float16* X3pad = (_Float16*)alloc((size_t)(N3 + 1) * 32 * 2);   // Kpad=32
    _Float16* Zpad  = (_Float16*)alloc((size_t)(N3 + 1) * 32 * 2);
    float*    Wcat  = (float*)alloc((size_t)27 * 16 * 10 * 4);
    _Float16* B1   = (_Float16*)alloc((size_t)27 * 3 * 6 * 512 * 2); // KB=3, NB=6
    _Float16* B1s  = (_Float16*)alloc((size_t)27 * 3 * 1 * 512 * 2); // KB=3, NB=1
    _Float16* B2   = (_Float16*)alloc((size_t)27 * 2 * 4 * 512 * 2); // KB=2, NB=4
    _Float16* B2s  = (_Float16*)alloc((size_t)27 * 2 * 1 * 512 * 2); // KB=2, NB=1
    _Float16* B3   = (_Float16*)alloc((size_t)27 * 1 * 2 * 512 * 2); // KB=1, NB=2
    _Float16* Bcat = (_Float16*)alloc((size_t)27 * 1 * 1 * 512 * 2); // KB=1, NB=1
    (void)ws_size; (void)n_in; (void)out_size;

    const int TB = 256;   // elementwise block
    const int WV = 8;     // waves per GEMM block
    const int MT = 2;     // row-tiles per wave
    const int TPB = WV * MT;  // row-tiles per block

    // --- Weight repacks ---
    {
        long t;
        t = (long)27 * 3 * 6 * 512;
        repack_weights<<<ceil_div(t, TB), TB, 0, stream>>>(W1, B1, 80, 80, 3, 6, t);
        t = (long)27 * 3 * 1 * 512;
        repack_weights<<<ceil_div(t, TB), TB, 0, stream>>>(W1s, B1s, 80, 2, 3, 1, t);
        t = (long)27 * 2 * 4 * 512;
        repack_weights<<<ceil_div(t, TB), TB, 0, stream>>>(W2, B2, 48, 48, 2, 4, t);
        t = (long)27 * 2 * 1 * 512;
        repack_weights<<<ceil_div(t, TB), TB, 0, stream>>>(W2s, B2s, 48, 2, 2, 1, t);
        t = (long)27 * 1 * 2 * 512;
        repack_weights<<<ceil_div(t, TB), TB, 0, stream>>>(W3, B3, 16, 16, 1, 2, t);
        concat_heads<<<ceil_div(27 * 16 * 10, TB), TB, 0, stream>>>(W3p, W3s, W3t, Wcat);
        t = (long)27 * 1 * 1 * 512;
        repack_weights<<<ceil_div(t, TB), TB, 0, stream>>>(Wcat, Bcat, 16, 10, 1, 1, t);
    }

    // --- Level 1 ---
    pad_features<<<ceil_div((long)(N1 + 1) * 96, TB), TB, 0, stream>>>(feat1, A1pad, N1, 80, 96);
    {
        int tiles = (N1 + 1 + 15) / 16;
        subconv_gemm<3, 6, MT, _Float16><<<ceil_div(tiles, TPB), WV * 32, 0, stream>>>(
            A1pad, nbr1, B1, Ypad, 96, 96, N1, N1 + 1);
        tiles = (N1 + 15) / 16;
        subconv_gemm<3, 1, MT, float><<<ceil_div(tiles, TPB), WV * 32, 0, stream>>>(
            Ypad, nbr1, B1s, s1, 2, 2, N1, N1);
    }
    mask_ppn<<<ceil_div(N1, TB), TB, 0, stream>>>(s1, coords1, mask1, out_ppn1, out_mask1, N1);

    // --- Level 2 ---
    gate_pad<<<ceil_div((long)(N2 + 1) * 64, TB), TB, 0, stream>>>(
        feat2, mask1, parent2, X2pad, N2, 48, 64);
    {
        int tiles = (N2 + 1 + 15) / 16;
        subconv_gemm<2, 4, MT, _Float16><<<ceil_div(tiles, TPB), WV * 32, 0, stream>>>(
            X2pad, nbr2, B2, Y2pad, 64, 64, N2, N2 + 1);
        tiles = (N2 + 15) / 16;
        subconv_gemm<2, 1, MT, float><<<ceil_div(tiles, TPB), WV * 32, 0, stream>>>(
            Y2pad, nbr2, B2s, s2, 2, 2, N2, N2);
    }
    mask_ppn<<<ceil_div(N2, TB), TB, 0, stream>>>(s2, coords2, mask2, out_ppn2, out_mask2, N2);

    // --- Level 3 ---
    gate_pad<<<ceil_div((long)(N3 + 1) * 32, TB), TB, 0, stream>>>(
        feat3, mask2, parent3, X3pad, N3, 16, 32);
    {
        int tiles = (N3 + 1 + 15) / 16;
        subconv_gemm<1, 2, MT, _Float16><<<ceil_div(tiles, TPB), WV * 32, 0, stream>>>(
            X3pad, nbr3, B3, Zpad, 32, 32, N3, N3 + 1);
        tiles = (N3 + 15) / 16;
        subconv_gemm<1, 1, MT, float><<<ceil_div(tiles, TPB), WV * 32, 0, stream>>>(
            Zpad, nbr3, Bcat, out_points, 10, 10, N3, N3);
    }
}
